// DecoderWithAttention_16114717295201
// MI455X (gfx1250) — matrix operand order repaired and checked by reading, not compiled, not run on hardware
//
#include <hip/hip_runtime.h>
#include <hip/hip_bf16.h>
#include <stdint.h>

// ---------------------------------------------------------------------------
// CDNA5 (gfx1250) LSTM decoder.
//   Phase 1: sort batch by length, convert weights/activations fp32->bf16.
//   Phase 2: Xg = enc_sorted @ W_ih^T + b_ih + b_hh   (parallel WMMA GEMM)
//   Phase 3: 128 sequential fused steps: gates = Xg[:,t] + h @ W_hh^T, LSTM.
//   Phase 4: preds = H @ W_fc^T + b_fc                (parallel WMMA GEMM)
// GEMMs: v_wmma_f32_16x16x32_bf16, 128x128 tile, double-buffered LDS filled
// by GLOBAL_LOAD_ASYNC_TO_LDS_B128 (ASYNCcnt-tracked DMA, zero data VGPRs).
// ---------------------------------------------------------------------------

typedef __attribute__((ext_vector_type(16))) __bf16 v16bf;
typedef __attribute__((ext_vector_type(8)))  float  v8f;

union FragBF { v16bf v; uint4 q[2]; };
union AccF   { v8f  v; float f[8]; };

#define BM 128
#define BN 128
#define BK 32

typedef __attribute__((address_space(3))) __bf16 lds_bf16_t;

// Async global->LDS 16-byte copy (per-lane addresses), tracked with ASYNCcnt.
__device__ __forceinline__ void async_b128(const __bf16* g, __bf16* l) {
  lds_bf16_t* lp = (lds_bf16_t*)l;          // addrspacecast: 32-bit LDS addr
  asm volatile("global_load_async_to_lds_b128 %0, %1, off"
               :: "v"(lp), "v"(g) : "memory");
}
__device__ __forceinline__ void wait_async0() {
  asm volatile("s_wait_asynccnt 0x0" ::: "memory");
}

// ---------------------------------------------------------------------------
// Stable descending argsort of B (=64) int64 lengths; also dec_len = len-1.
// ---------------------------------------------------------------------------
__global__ void k_sort(const long long* __restrict__ lens,
                       int* __restrict__ sort_ind,
                       int* __restrict__ dec_len, int B) {
  __shared__ long long sl[64];
  int i = threadIdx.x;
  if (i < B) sl[i] = lens[i];
  __syncthreads();
  if (i < B) {
    long long me = sl[i];
    int rank = 0;
    for (int j = 0; j < B; ++j) {
      long long v = sl[j];
      if (v > me || (v == me && j < i)) ++rank;   // stable, descending
    }
    sort_ind[rank] = i;
    dec_len[rank] = (int)(me - 1);
  }
}

__global__ __launch_bounds__(256)
void k_f32_to_bf16(const float* __restrict__ src, __bf16* __restrict__ dst,
                   size_t n) {
  size_t i = (size_t)blockIdx.x * 256 + threadIdx.x;
  if (i < n) dst[i] = (__bf16)src[i];
}

// Gather encoder rows in sorted order and convert to bf16.
__global__ __launch_bounds__(256)
void k_gather_enc(const float* __restrict__ enc, const int* __restrict__ sort_ind,
                  __bf16* __restrict__ dst, int T, int E) {
  int b = blockIdx.y, t = blockIdx.x;
  const float* src = &enc[((size_t)sort_ind[b] * T + t) * E];
  __bf16* d = &dst[((size_t)b * T + t) * E];
  for (int e = threadIdx.x; e < E; e += 256) d[e] = (__bf16)src[e];
}

__global__ __launch_bounds__(256)
void k_zero(uint32_t* __restrict__ p, size_t n) {
  size_t i = (size_t)blockIdx.x * 256 + threadIdx.x;
  if (i < n) p[i] = 0u;
}

// ---------------------------------------------------------------------------
// bf16 WMMA GEMM:  C[MxN] = A[MxK] * B[NxK]^T + bias1 (+ bias2)
//   A row-major bf16, B row-major bf16 (weight layout (N,K) == B^T), C fp32.
//   M, K multiples of 128/32; N arbitrary (B rows clamped, stores guarded).
//   Block: 256 thr (8 waves) -> 128x128 tile; wave grid 4(M) x 2(N);
//   each wave: 2 M-subtiles x 4 N-subtiles = 8 accumulators.
//   Double-buffered LDS filled by async DMA issued before the WMMA block.
// ---------------------------------------------------------------------------
__global__ __launch_bounds__(256)
void k_gemm_bf16(const __bf16* __restrict__ A, const __bf16* __restrict__ Bw,
                 const float* __restrict__ bias1, const float* __restrict__ bias2,
                 float* __restrict__ C, int M, int N, int K) {
  __shared__ __bf16 lA[2][BM * BK];   // 2 x 8 KB
  __shared__ __bf16 lB[2][BN * BK];   // 2 x 8 KB
  const int tid  = threadIdx.x;
  const int wave = tid >> 5;
  const int lane = tid & 31;
  const int lm   = lane & 15;   // row-in-subtile (A/B) / column (C)
  const int lh   = lane >> 4;   // half select
  const int wm   = wave & 3;    // M strip (32 rows)
  const int wn   = wave >> 2;   // N strip (64 cols)
  const int m0 = blockIdx.x * BM;
  const int n0 = blockIdx.y * BN;

  // Staging coordinates: 512 x 16B per tile, 2 per thread.
  const int sr0 = (tid + 0)   >> 2, sc0 = ((tid + 0)   & 3) * 8;
  const int sr1 = (tid + 256) >> 2, sc1 = ((tid + 256) & 3) * 8;
  // Clamp OOB B rows to a valid row: those columns are never stored.
  const int bq0 = (n0 + sr0 < N) ? n0 + sr0 : N - 1;
  const int bq1 = (n0 + sr1 < N) ? n0 + sr1 : N - 1;

  const v8f vzero = {0.f, 0.f, 0.f, 0.f, 0.f, 0.f, 0.f, 0.f};
  v8f acc[2][4] = {{vzero, vzero, vzero, vzero}, {vzero, vzero, vzero, vzero}};

  auto stage = [&](int buf, int k0) {
    async_b128(&A[(size_t)(m0 + sr0) * K + k0 + sc0], &lA[buf][sr0 * BK + sc0]);
    async_b128(&A[(size_t)(m0 + sr1) * K + k0 + sc1], &lA[buf][sr1 * BK + sc1]);
    async_b128(&Bw[(size_t)bq0 * K + k0 + sc0],       &lB[buf][sr0 * BK + sc0]);
    async_b128(&Bw[(size_t)bq1 * K + k0 + sc1],       &lB[buf][sr1 * BK + sc1]);
  };

  const int nk = K / BK;
  stage(0, 0);
  wait_async0();
  __syncthreads();

  for (int k = 0; k < nk; ++k) {
    const int buf = k & 1;
    if (k + 1 < nk)                  // DMA next chunk under this chunk's WMMAs
      stage(buf ^ 1, (k + 1) * BK);  // safe: barrier @k-1 ended reads of buf^1

    // A fragments for this wave's two 16-row subtiles.
    FragBF fa[2];
#pragma unroll
    for (int mt = 0; mt < 2; ++mt) {
      const int ar = wm * 32 + mt * 16 + lm;
      fa[mt].q[0] = *(const uint4*)&lA[buf][ar * BK + lh * 8];
      fa[mt].q[1] = *(const uint4*)&lA[buf][ar * BK + 16 + lh * 8];
    }
#pragma unroll
    for (int nt = 0; nt < 4; ++nt) {
      FragBF fb;
      const int br = wn * 64 + nt * 16 + lm;
      fb.q[0] = *(const uint4*)&lB[buf][br * BK + lh * 8];
      fb.q[1] = *(const uint4*)&lB[buf][br * BK + 16 + lh * 8];
#pragma unroll
      for (int mt = 0; mt < 2; ++mt)
        acc[mt][nt] = __builtin_amdgcn_wmma_f32_16x16x32_bf16(
            false, fa[mt].v, false, fb.v, (short)0, acc[mt][nt], false, false);
    }
    if (k + 1 < nk) {
      wait_async0();                 // my async writes to buf^1 have landed
      __syncthreads();               // everyone's landed; reads of buf done
    }
  }

  // Epilogue: bias add + store. C/D layout: VGPR j -> M=j (+8 hi half), lane lm -> col.
#pragma unroll
  for (int nt = 0; nt < 4; ++nt) {
    int n = n0 + wn * 64 + nt * 16 + lm;
    if (n >= N) continue;
    float b = (bias1 ? bias1[n] : 0.f) + (bias2 ? bias2[n] : 0.f);
#pragma unroll
    for (int mt = 0; mt < 2; ++mt) {
      AccF a; a.v = acc[mt][nt];
#pragma unroll
      for (int j = 0; j < 8; ++j) {
        int m = m0 + wm * 32 + mt * 16 + ((lane < 16) ? j : j + 8);
        C[(size_t)m * N + n] = a.f[j] + b;
      }
    }
  }
}

// ---------------------------------------------------------------------------
// One fused LSTM step. Grid: D/16 = 64 blocks; block owns column slice
// [n0, n0+16) of EVERY gate, so the pointwise LSTM can be fused in-kernel.
// 8 waves x 2 tiles = 16 (gate, mtile) 16x16 WMMA tiles (M=64, 4 gates).
// gates = h_prev_bf @ W_hh^T + Xg[:, t];  then i,f,g,o -> c,h (masked).
// ---------------------------------------------------------------------------
__global__ __launch_bounds__(256)
void k_lstm_step(const __bf16* __restrict__ hprev,   // 64 x D bf16 (t-1)
                 const __bf16* __restrict__ Whh,     // 4D x D bf16
                 const float*  __restrict__ Xg,      // (B*T) x 4D, row b*T+t
                 const int*    __restrict__ dec_len, // 64
                 float*  __restrict__ cstate,        // 64 x D f32 (in-place)
                 float*  __restrict__ hstate,        // 64 x D f32 (in-place)
                 __bf16* __restrict__ hnext,         // 64 x D bf16 (t)
                 __bf16* __restrict__ Hbf,           // (B*T) x D bf16
                 int t, int T, int D) {
  __shared__ float lg[4][64][16];   // 16 KB gate buffer
  const int tid  = threadIdx.x;
  const int wave = tid >> 5;
  const int lane = tid & 31;
  const int lm   = lane & 15;
  const int lh   = lane >> 4;
  const int n0   = blockIdx.x * 16;
  const int gate = (2 * wave) >> 2;           // both tiles of a wave share gate

  const v8f vzero = {0.f, 0.f, 0.f, 0.f, 0.f, 0.f, 0.f, 0.f};
  v8f acc[2] = {vzero, vzero};

  for (int k0 = 0; k0 < D; k0 += BK) {
    FragBF fb;                                 // W_hh row = gate*D + n0 + lm
    const __bf16* brow = &Whh[(size_t)(gate * D + n0 + lm) * D + k0];
    fb.q[0] = *(const uint4*)&brow[lh * 8];
    fb.q[1] = *(const uint4*)&brow[16 + lh * 8];
#pragma unroll
    for (int i = 0; i < 2; ++i) {
      const int mt = (2 * wave + i) & 3;
      FragBF fa;
      const __bf16* arow = &hprev[(size_t)(mt * 16 + lm) * D + k0];
      fa.q[0] = *(const uint4*)&arow[lh * 8];
      fa.q[1] = *(const uint4*)&arow[16 + lh * 8];
      acc[i] = __builtin_amdgcn_wmma_f32_16x16x32_bf16(
          false, fa.v, false, fb.v, (short)0, acc[i], false, false);
    }
  }

  // Spill gate accumulators to LDS so any thread can read all 4 gates.
#pragma unroll
  for (int i = 0; i < 2; ++i) {
    const int tile = 2 * wave + i, g = tile >> 2, mt = tile & 3;
    AccF a; a.v = acc[i];
#pragma unroll
    for (int j = 0; j < 8; ++j) {
      int m = mt * 16 + ((lane < 16) ? j : j + 8);
      lg[g][m][lm] = a.f[j];
    }
  }
  __syncthreads();

  // Pointwise LSTM on 64 rows x 16 cols (4 elements / thread), fp32.
  for (int e = tid; e < 64 * 16; e += 256) {
    int m = e >> 4, j = e & 15;
    int col = n0 + j;
    const float* xg = &Xg[((size_t)m * T + t) * 4 * D];
    float gi = lg[0][m][j] + xg[0 * D + col];
    float gf = lg[1][m][j] + xg[1 * D + col];
    float gg = lg[2][m][j] + xg[2 * D + col];
    float go = lg[3][m][j] + xg[3 * D + col];
    gi = 1.f / (1.f + __expf(-gi));
    gf = 1.f / (1.f + __expf(-gf));
    gg = tanhf(gg);
    go = 1.f / (1.f + __expf(-go));
    size_t idx = (size_t)m * D + col;
    float c_old = cstate[idx];
    float h_old = hstate[idx];
    float c_new = gf * c_old + gi * gg;
    float h_new = go * tanhf(c_new);
    bool active = t < dec_len[m];
    float c_out = active ? c_new : c_old;
    float h_out = active ? h_new : h_old;
    cstate[idx] = c_out;
    hstate[idx] = h_out;
    hnext[idx]  = (__bf16)h_out;
    Hbf[((size_t)m * T + t) * D + col] = (__bf16)h_out;
  }
}

// ---------------------------------------------------------------------------
// Host side.
// ---------------------------------------------------------------------------
static inline size_t align256(size_t x) { return (x + 255) & ~(size_t)255; }

extern "C" void kernel_launch(void* const* d_in, const int* in_sizes, int n_in,
                              void* d_out, int out_size, void* d_ws, size_t ws_size,
                              hipStream_t stream) {
  const float*     enc  = (const float*)d_in[0];
  const long long* lens = (const long long*)d_in[1];   // int64 lengths
  const float*     W_ih = (const float*)d_in[2];
  const float*     b_ih = (const float*)d_in[3];
  const float*     W_hh = (const float*)d_in[4];
  const float*     b_hh = (const float*)d_in[5];
  const float*     W_fc = (const float*)d_in[6];
  const float*     b_fc = (const float*)d_in[7];
  float* out = (float*)d_out;

  const int B = in_sizes[1];                 // 64
  const int D = in_sizes[3] / 4;             // 1024
  const int E = in_sizes[2] / (4 * D);       // 1024
  const int T = in_sizes[0] / (B * E);       // 128
  const int V = in_sizes[7];                 // 10000
  const int M = B * T;                       // 8192

  // ---- workspace layout ----
  char* w = (char*)d_ws;
  size_t off = 0;
  auto bump = [&](size_t bytes) { void* p = w + off; off = align256(off + bytes); return p; };
  int*    sort_ind = (int*)bump(B * sizeof(int));
  int*    dec_len  = (int*)bump(B * sizeof(int));
  __bf16* Wih_bf   = (__bf16*)bump((size_t)4 * D * E * 2);
  __bf16* Whh_bf   = (__bf16*)bump((size_t)4 * D * D * 2);
  __bf16* Wfc_bf   = (__bf16*)bump((size_t)V * D * 2);
  __bf16* Xbf      = (__bf16*)bump((size_t)M * E * 2);
  float*  Xg       = (float*)bump((size_t)M * 4 * D * 4);
  __bf16* Hbf      = (__bf16*)bump((size_t)M * D * 2);
  __bf16* hA       = (__bf16*)bump((size_t)B * D * 2);
  __bf16* hB       = (__bf16*)bump((size_t)B * D * 2);
  float*  cstate   = (float*)bump((size_t)B * D * 4);
  float*  hstate   = (float*)bump((size_t)B * D * 4);
  (void)ws_size; (void)n_in; (void)out_size;

  // ---- phase 0: sort + conversions + state init ----
  k_sort<<<1, 64, 0, stream>>>(lens, sort_ind, dec_len, B);

  {
    size_t n1 = (size_t)4 * D * E;
    size_t n2 = (size_t)4 * D * D;
    size_t n3 = (size_t)V * D;
    k_f32_to_bf16<<<(unsigned)((n1 + 255) / 256), 256, 0, stream>>>(W_ih, Wih_bf, n1);
    k_f32_to_bf16<<<(unsigned)((n2 + 255) / 256), 256, 0, stream>>>(W_hh, Whh_bf, n2);
    k_f32_to_bf16<<<(unsigned)((n3 + 255) / 256), 256, 0, stream>>>(W_fc, Wfc_bf, n3);
  }
  k_gather_enc<<<dim3(T, B), 256, 0, stream>>>(enc, sort_ind, Xbf, T, E);

  {
    // zero hA, hB, cstate, hstate (contiguous region)
    size_t zb = ((size_t)B * D * 2) * 2 + ((size_t)B * D * 4) * 2;
    size_t zn = (zb + 3) / 4;
    k_zero<<<(unsigned)((zn + 255) / 256), 256, 0, stream>>>((uint32_t*)hA, zn);
  }

  // ---- phase 1: Xg = Xbf @ W_ih^T + b_ih + b_hh   (M x 4D) ----
  k_gemm_bf16<<<dim3(M / BM, (4 * D + BN - 1) / BN), 256, 0, stream>>>(
      Xbf, Wih_bf, b_ih, b_hh, Xg, M, 4 * D, E);

  // ---- phase 2: 128 sequential fused LSTM steps (ping-pong h) ----
  for (int t = 0; t < T; ++t) {
    const __bf16* hp = (t & 1) ? hB : hA;
    __bf16*       hn = (t & 1) ? hA : hB;
    k_lstm_step<<<D / 16, 256, 0, stream>>>(hp, Whh_bf, Xg, dec_len,
                                            cstate, hstate, hn, Hbf, t, T, D);
  }

  // ---- phase 3: preds = Hbf @ W_fc^T + b_fc   (M x V) -> d_out ----
  k_gemm_bf16<<<dim3(M / BM, (V + BN - 1) / BN), 256, 0, stream>>>(
      Hbf, Wfc_bf, b_fc, nullptr, out, M, V, D);
}
